// LSTM_17695265259585
// MI455X (gfx1250) — matrix operand order repaired and checked by reading, not compiled, not run on hardware
//
#include <hip/hip_runtime.h>
#include <math.h>

#define SEQ   8192
#define INSZ  620
#define KPAD  640          // K padded to 20 chunks of 32
#define HID   1000
#define G4    4000
#define MT    (SEQ / 16)   // 512 tiles in M
#define NT    (G4  / 16)   // 250 tiles in N
#define KCH   (KPAD / 32)  // 20 K-chunks
#define WPB   8            // waves per block (256 threads, wave32)
#define NB    125          // recurrence blocks (grid-resident)
#define UPB   8            // hidden units per block (125*8 = 1000)

typedef __attribute__((ext_vector_type(16))) __bf16 v16bf;
typedef __attribute__((ext_vector_type(8)))  __bf16 v8bf;
typedef __attribute__((ext_vector_type(8)))  float  v8f;

static __device__ __forceinline__ __bf16 f2bf(float f) {
    unsigned u = __builtin_bit_cast(unsigned, f);
    u += 0x7FFFu + ((u >> 16) & 1u);          // round-to-nearest-even
    unsigned short h = (unsigned short)(u >> 16);
    return __builtin_bit_cast(__bf16, h);
}

// ---------------------------------------------------------------------------
// Prep: zero-padded bf16 copies.  dst is rows x KPAD, src is rows x INSZ.
// ---------------------------------------------------------------------------
__global__ __launch_bounds__(256)
void lstm_pad_bf16(const float* __restrict__ src, __bf16* __restrict__ dst,
                   int rows) {
    const size_t n = (size_t)rows * KPAD;
    for (size_t i = (size_t)blockIdx.x * 256 + threadIdx.x; i < n;
         i += (size_t)gridDim.x * 256) {
        const int r = (int)(i / KPAD);
        const int k = (int)(i % KPAD);
        const float v = (k < INSZ) ? src[(size_t)r * INSZ + k] : 0.f;
        dst[i] = f2bf(v);
    }
}

// ---------------------------------------------------------------------------
// Phase 1: xg[t, n] = sum_k x[t,k] * W_ih[n,k] + (b_ih[n] + b_hh[n])
// One 16x16 tile per wave; branch-free b128 fragment loads from padded bf16.
// ---------------------------------------------------------------------------
__global__ __launch_bounds__(256)
void lstm_xg_wmma(const __bf16* __restrict__ xb, const __bf16* __restrict__ wb,
                  const float* __restrict__ bih, const float* __restrict__ bhh,
                  float* __restrict__ xg) {
    const int lane = threadIdx.x & 31;
    const int wid  = threadIdx.x >> 5;
    const int tile = blockIdx.x * WPB + wid;          // grid sized exactly MT*NT
    const int tm = tile / NT;
    const int tn = tile % NT;

    const int rowA = tm * 16 + (lane & 15);           // A: one row per lane
    const int colB = tn * 16 + (lane & 15);           // B/C/D: one col per lane
    const __bf16* __restrict__ xr = xb + (size_t)rowA * KPAD;
    const __bf16* __restrict__ wr = wb + (size_t)colB * KPAD;

    // Contiguous 16-bit fragment K bases, k = 16*(lane/16) + s for A and B (wave32)
    const int kbA_lo = (lane < 16) ? 0  : 16;         // A VGPR 0..3
    const int kbA_hi = (lane < 16) ? 8  : 24;         // A VGPR 4..7
    const int kbB    = (lane < 16) ? 0  : 16;         // B VGPR 0..7

    v8f acc = {};
#pragma unroll 4
    for (int kc = 0; kc < KCH; ++kc) {
        const int k0 = kc * 32;
        const v8bf alo = *(const v8bf*)(xr + k0 + kbA_lo);   // 16B, aligned
        const v8bf ahi = *(const v8bf*)(xr + k0 + kbA_hi);   // 16B, aligned
        const v16bf a = __builtin_shufflevector(alo, ahi,
            0, 1, 2, 3, 4, 5, 6, 7, 8, 9, 10, 11, 12, 13, 14, 15);
        const v16bf b = *(const v16bf*)(wr + k0 + kbB);      // 32B, aligned
        acc = __builtin_amdgcn_wmma_f32_16x16x32_bf16(
            /*neg_a=*/false, a, /*neg_b=*/false, b,
            /*c_mod=*/(short)0, acc, /*reuse_a=*/false, /*reuse_b=*/false);
    }

    const float bias = bih[colB] + bhh[colB];
#pragma unroll
    for (int r = 0; r < 8; ++r) {
        const int m = tm * 16 + ((lane < 16) ? r : (r + 8));  // C/D layout
        xg[(size_t)m * G4 + colB] = acc[r] + bias;
    }
}

// ---------------------------------------------------------------------------
// Init: zero h double-buffer and the grid-barrier counter (graph-replay safe).
// ---------------------------------------------------------------------------
__global__ void lstm_init(float* hbuf, unsigned* bar) {
    for (int i = threadIdx.x; i < 2048; i += 256) hbuf[i] = 0.f;
    if (threadIdx.x == 0) *bar = 0u;
}

// ---------------------------------------------------------------------------
// Phase 2: persistent grid-resident recurrence. Block b owns units
// [b*UPB, (b+1)*UPB): all 4 gate rows for those units -> no cross-block data
// needed for the cell update. One wave reduces one 1000-long dot product.
// One device-wide barrier per timestep (monotonic counter).
// ---------------------------------------------------------------------------
__global__ __launch_bounds__(256)
void lstm_rec(const float* __restrict__ xg, const float* __restrict__ Whh,
              float* __restrict__ out, float* __restrict__ hbuf,
              unsigned* __restrict__ bar) {
    __shared__ float sh[1024];   // h_{t-1}
    __shared__ float sg[32];     // 4 gates x UPB pre-activations

    const int tid  = threadIdx.x;
    const int wid  = tid >> 5;
    const int lane = tid & 31;
    const int u0   = blockIdx.x * UPB;

    float c = 0.f;               // cell state, live in threads 0..UPB-1

    for (int t = 0; t < SEQ; ++t) {
        const float* hprev = hbuf + (t & 1) * 1024;
        for (int k = tid; k < HID; k += 256) sh[k] = hprev[k];
        __syncthreads();

        // 32 rows (4 gates x 8 units), 8 waves -> 4 passes, 1 row per wave.
#pragma unroll
        for (int p = 0; p < 4; ++p) {
            const int ri   = p * 8 + wid;            // 0..31
            const int gate = ri >> 3;                // 0..3 (i,f,g,o)
            const int ul   = ri & 7;
            const float* __restrict__ w = Whh + (size_t)(gate * HID + u0 + ul) * HID;
            __builtin_prefetch(w + lane * 4, 0, 0);  // global_prefetch_b8
            float acc = 0.f;
            for (int k = lane; k < HID; k += 32)
                acc = fmaf(w[k], sh[k], acc);
#pragma unroll
            for (int off = 16; off > 0; off >>= 1)
                acc += __shfl_down(acc, off, 32);
            if (lane == 0) sg[ri] = acc;
        }
        __syncthreads();

        if (tid < UPB) {
            const int u = u0 + tid;
            const float* xgt = xg + (size_t)t * G4;
            const float gi = sg[ 0 + tid] + xgt[0 * HID + u];
            const float gf = sg[ 8 + tid] + xgt[1 * HID + u];
            const float gg = sg[16 + tid] + xgt[2 * HID + u];
            const float go = sg[24 + tid] + xgt[3 * HID + u];
            const float i = 1.f / (1.f + __expf(-gi));
            const float f = 1.f / (1.f + __expf(-gf));
            const float g = tanhf(gg);
            const float o = 1.f / (1.f + __expf(-go));
            c = f * c + i * g;
            const float h = o * tanhf(c);
            hbuf[((t + 1) & 1) * 1024 + u] = h;
            out[(size_t)t * HID + u]       = h;
        }
        __syncthreads();

        // Device-wide barrier: monotonic counter, target NB*(t+1).
        if (tid == 0) {
            __threadfence();
            atomicAdd(bar, 1u);
            const unsigned target = (unsigned)(t + 1) * (unsigned)NB;
            while (__hip_atomic_load(bar, __ATOMIC_RELAXED,
                                     __HIP_MEMORY_SCOPE_AGENT) < target) {
                __builtin_amdgcn_s_sleep(1);
            }
            __threadfence();
        }
        __syncthreads();
    }
}

// ---------------------------------------------------------------------------
extern "C" void kernel_launch(void* const* d_in, const int* in_sizes, int n_in,
                              void* d_out, int out_size, void* d_ws, size_t ws_size,
                              hipStream_t stream) {
    const float* x   = (const float*)d_in[0];
    const float* Wih = (const float*)d_in[1];
    const float* Whh = (const float*)d_in[2];
    const float* bih = (const float*)d_in[3];
    const float* bhh = (const float*)d_in[4];
    float* out = (float*)d_out;

    // Scratch layout (256 B aligned regions)
    char* ws = (char*)d_ws;
    float*   xg = (float*)ws;                               // SEQ*G4 f32 (131 MB)
    size_t off = (size_t)SEQ * G4 * sizeof(float);
    __bf16* xb = (__bf16*)(ws + off);                       // SEQ*KPAD bf16
    off += (size_t)SEQ * KPAD * sizeof(__bf16);
    __bf16* wb = (__bf16*)(ws + off);                       // G4*KPAD bf16
    off += (size_t)G4 * KPAD * sizeof(__bf16);
    float* hbuf = (float*)(ws + off);                       // 2 x 1024 f32
    off += 2048 * sizeof(float);
    unsigned* bar = (unsigned*)(ws + off);                  // barrier counter

    lstm_init<<<1, 256, 0, stream>>>(hbuf, bar);
    lstm_pad_bf16<<<4096, 256, 0, stream>>>(x,   xb, SEQ);
    lstm_pad_bf16<<<4096, 256, 0, stream>>>(Wih, wb, G4);
    lstm_xg_wmma<<<(MT * NT) / WPB, 256, 0, stream>>>(xb, wb, bih, bhh, xg);
    lstm_rec<<<NB, 256, 0, stream>>>(xg, Whh, out, hbuf, bar);
}
